// GATModel_43267500540824
// MI455X (gfx1250) — compile-verified, matrix operand order checked by reading, not run on hardware
//
#include <hip/hip_runtime.h>

typedef __attribute__((ext_vector_type(16))) __bf16 v16bf;
typedef __attribute__((ext_vector_type(8)))  __bf16 v8bf;
typedef __attribute__((ext_vector_type(8)))  float  v8f;

// ---------------------------------------------------------------------------
// Utility kernels
// ---------------------------------------------------------------------------
__global__ __launch_bounds__(256) void k_convert_bf16(const float* __restrict__ in,
                                                      __bf16* __restrict__ out, int n) {
  int i = blockIdx.x * 256 + threadIdx.x;
  if (i < n) out[i] = (__bf16)in[i];
}

__global__ __launch_bounds__(256) void k_fill_f32(float* __restrict__ p, float v, int n) {
  int i = blockIdx.x * 256 + threadIdx.x;
  if (i < n) p[i] = v;
}

__global__ __launch_bounds__(256) void k_init_bias(float* __restrict__ out,
                                                   const float* __restrict__ b, int n, int Cc) {
  int i = blockIdx.x * 256 + threadIdx.x;
  if (i < n) out[i] = b[i % Cc];
}

// ---------------------------------------------------------------------------
// Pre-pack weights [K, Nc] f32 (row-major) into WMMA B-fragment order, bf16.
// Chunk cb = kb*(Nc/16)+nb holds a 32(K) x 16(N) fragment: 32 lanes x 16 elems.
// ISA 16-bit layout assumption: lane L -> col = nb*16 + L%16;
// element i -> K = kb*32 + (i%8) + 16*(i/8) + 8*(L/16).
// One thread produces one lane's 16 contiguous bf16 (32B store).
// ---------------------------------------------------------------------------
__global__ __launch_bounds__(256) void k_prepack_b(const float* __restrict__ W,
                                                   __bf16* __restrict__ out,
                                                   int K, int Nc) {
  int t = blockIdx.x * 256 + threadIdx.x;        // (cb, lane)
  int total = (K >> 5) * (Nc >> 4) * 32;
  if (t >= total) return;
  int L  = t & 31;
  int cb = t >> 5;
  int nbPerRow = Nc >> 4;
  int nb = cb % nbPerRow;
  int kb = cb / nbPerRow;
  int col   = nb * 16 + (L & 15);
  int kbase = kb * 32 + ((L >> 4) << 3);
  v16bf v;
  #pragma unroll
  for (int i = 0; i < 16; ++i) {
    int k = kbase + (i & 7) + ((i >> 3) << 4);
    v[i] = (__bf16)W[(size_t)k * Nc + col];
  }
  *(v16bf*)(out + (size_t)t * 16) = v;
}

// ---------------------------------------------------------------------------
// Barrier-free WMMA bf16 GEMM, operands streamed straight from L2.
// C[M,Nc] = A[M,K] * Bp (pre-packed), fp32 accumulate.
// 256 threads = 8 waves in a 4x2 grid over a 128 x BN block tile.
// Wave tile: 32 x (BN/2)  ->  2 x JF fragments, JF = BN/32.
// A fragment = two 16B global loads per lane (row-major K-runs).
// B fragment = one 32B contiguous read from the pre-packed panel.
// ---------------------------------------------------------------------------
template <int BN>   // 128 (layer 1) or 64 (layer 2)
__global__ __launch_bounds__(256) void k_wmma_gemm_direct(
    const __bf16* __restrict__ A,    // [M,K] row-major bf16
    const __bf16* __restrict__ Bp,   // pre-packed fragments
    float* __restrict__ C,           // [M,Nc] f32
    int M, int K, int Nc) {
  constexpr int JF = BN / 32;        // N-fragments per wave
  const int tid  = threadIdx.x;
  const int lane = tid & 31;
  const int wave = tid >> 5;
  const int wm   = wave >> 1;        // 0..3 : 32-row group
  const int wn   = wave & 1;         // 0..1 : (BN/2)-col group
  const int rowBase = blockIdx.y * 128;
  const int colBase = blockIdx.x * BN;
  const int nbPerRow = Nc >> 4;
  const int nbBase = (colBase >> 4) + wn * JF;

  v8f acc[2][JF] = {};

  // per-lane A addresses (rows clamped; C stores are guarded, padded rows only
  // pollute outputs we never write — EXEC stays all-ones for WMMA)
  int r0 = rowBase + wm * 32 + (lane & 15);
  int r1 = r0 + 16;
  if (r0 >= M) r0 = M - 1;
  if (r1 >= M) r1 = M - 1;
  const __bf16* a0 = A + (size_t)r0 * K + ((lane >> 4) << 3);
  const __bf16* a1 = A + (size_t)r1 * K + ((lane >> 4) << 3);

  for (int k0 = 0; k0 < K; k0 += 32) {
    v16bf af[2];
    {
      v8bf lo0 = *(const v8bf*)(a0 + k0);
      v8bf hi0 = *(const v8bf*)(a0 + k0 + 16);
      v8bf lo1 = *(const v8bf*)(a1 + k0);
      v8bf hi1 = *(const v8bf*)(a1 + k0 + 16);
      #pragma unroll
      for (int t2 = 0; t2 < 8; ++t2) {
        af[0][t2] = lo0[t2]; af[0][t2 + 8] = hi0[t2];
        af[1][t2] = lo1[t2]; af[1][t2 + 8] = hi1[t2];
      }
    }
    const int kb = k0 >> 5;
    #pragma unroll
    for (int j = 0; j < JF; ++j) {
      const v16bf bf =
          *(const v16bf*)(Bp + ((size_t)(kb * nbPerRow + nbBase + j) * 32 + lane) * 16);
      acc[0][j] = __builtin_amdgcn_wmma_f32_16x16x32_bf16(
          false, af[0], false, bf, (short)0, acc[0][j], false, false);
      acc[1][j] = __builtin_amdgcn_wmma_f32_16x16x32_bf16(
          false, af[1], false, bf, (short)0, acc[1][j], false, false);
    }
    if (k0 + 32 < K) {                 // near-cache prefetch of next A runs
      __builtin_prefetch(a0 + k0 + 32, 0, 3);
      __builtin_prefetch(a1 + k0 + 32, 0, 3);
    }
  }

  // store C: lane -> N = lane%16; VGPR r -> M = r + 8*(lane/16)
  #pragma unroll
  for (int i = 0; i < 2; ++i) {
    int rbase = rowBase + wm * 32 + i * 16 + ((lane >> 4) << 3);
    #pragma unroll
    for (int j = 0; j < JF; ++j) {
      int col = colBase + (wn * JF + j) * 16 + (lane & 15);
      #pragma unroll
      for (int r = 0; r < 8; ++r) {
        int row = rbase + r;
        if (row < M) C[(size_t)row * Nc + col] = acc[i][j][r];
      }
    }
  }
}

// ---------------------------------------------------------------------------
// Attention coefficient dots: one wave per (node, head), C = 64.
// ---------------------------------------------------------------------------
__global__ __launch_bounds__(256) void k_attn(const float* __restrict__ h,
                                              const float* __restrict__ aSrc,
                                              const float* __restrict__ aDst,
                                              float* __restrict__ oSrc,
                                              float* __restrict__ oDst,
                                              int N, int H) {
  int wid  = blockIdx.x * 8 + (threadIdx.x >> 5);
  int lane = threadIdx.x & 31;
  if (wid >= N * H) return;
  int hh = wid % H;
  const float* row = h + (size_t)wid * 64;
  float x0 = row[lane], x1 = row[lane + 32];
  const float* s = aSrc + hh * 64;
  const float* d = aDst + hh * 64;
  float vs = x0 * s[lane] + x1 * s[lane + 32];
  float vd = x0 * d[lane] + x1 * d[lane + 32];
  #pragma unroll
  for (int off = 16; off > 0; off >>= 1) {
    vs += __shfl_xor(vs, off, 32);
    vd += __shfl_xor(vd, off, 32);
  }
  if (lane == 0) { oSrc[wid] = vs; oDst[wid] = vd; }
}

// ---------------------------------------------------------------------------
// Edge kernels (segment softmax + aggregation). Self-loops are edges >= E.
// ---------------------------------------------------------------------------
__device__ __forceinline__ void edgeSrcDst(const long long* __restrict__ ei, int E,
                                           int e, int& s, int& d) {
  if (e < E) { s = (int)ei[e]; d = (int)ei[E + e]; }
  else       { s = e - E; d = s; }
}

__device__ __forceinline__ void atomicMaxF(float* addr, float v) {
  if (v >= 0.f) atomicMax((int*)addr, __float_as_int(v));
  else          atomicMin((unsigned int*)addr, (unsigned int)__float_as_int(v));
}

__global__ __launch_bounds__(256) void k_edge_max(const long long* __restrict__ ei,
                                                  int E, int N, int H,
                                                  const float* __restrict__ aS,
                                                  const float* __restrict__ aD,
                                                  float* __restrict__ m) {
  int gid = blockIdx.x * 256 + threadIdx.x;
  if (gid >= (E + N) * H) return;
  int e = gid / H, hh = gid - e * H;
  int s, d; edgeSrcDst(ei, E, e, s, d);
  float v = aS[s * H + hh] + aD[d * H + hh];
  v = v > 0.f ? v : 0.2f * v;                 // leaky_relu, slope 0.2
  atomicMaxF(&m[d * H + hh], v);
}

__global__ __launch_bounds__(256) void k_edge_expsum(const long long* __restrict__ ei,
                                                     int E, int N, int H,
                                                     const float* __restrict__ aS,
                                                     const float* __restrict__ aD,
                                                     const float* __restrict__ m,
                                                     float* __restrict__ alphaE,
                                                     float* __restrict__ ssum) {
  int gid = blockIdx.x * 256 + threadIdx.x;
  if (gid >= (E + N) * H) return;
  int e = gid / H, hh = gid - e * H;
  int s, d; edgeSrcDst(ei, E, e, s, d);
  float v = aS[s * H + hh] + aD[d * H + hh];
  v = v > 0.f ? v : 0.2f * v;
  float ex = __expf(v - m[d * H + hh]);
  alphaE[gid] = ex;
  atomicAdd(&ssum[d * H + hh], ex);
}

__global__ __launch_bounds__(256) void k_edge_agg(const long long* __restrict__ ei,
                                                  int E, int N, int H,
                                                  const float* __restrict__ alphaE,
                                                  const float* __restrict__ ssum,
                                                  const float* __restrict__ h,
                                                  float* __restrict__ out) {
  int gid = blockIdx.x * 256 + threadIdx.x;   // 64 threads per edge
  int e = gid >> 6;
  if (e >= E + N) return;
  int c = gid & 63;
  int s, d; edgeSrcDst(ei, E, e, s, d);
  for (int hh = 0; hh < H; ++hh) {
    float a = alphaE[e * H + hh] / (ssum[d * H + hh] + 1e-16f);
    int col = hh * 64 + c;
    atomicAdd(&out[(size_t)d * (H * 64) + col], a * h[(size_t)s * (H * 64) + col]);
  }
}

__global__ __launch_bounds__(256) void k_relu_bias_bf16(const float* __restrict__ in,
                                                        const float* __restrict__ b,
                                                        __bf16* __restrict__ out,
                                                        int n, int F) {
  int i = blockIdx.x * 256 + threadIdx.x;
  if (i < n) {
    float v = in[i] + b[i % F];
    out[i] = (__bf16)fmaxf(v, 0.f);
  }
}

// ---------------------------------------------------------------------------
// Launch
// ---------------------------------------------------------------------------
extern "C" void kernel_launch(void* const* d_in, const int* in_sizes, int n_in,
                              void* d_out, int out_size, void* d_ws, size_t ws_size,
                              hipStream_t stream) {
  const float*     x    = (const float*)d_in[0];
  const long long* ei   = (const long long*)d_in[1];
  const float*     W1   = (const float*)d_in[2];
  const float*     aS1  = (const float*)d_in[3];
  const float*     aD1  = (const float*)d_in[4];
  const float*     b1   = (const float*)d_in[5];
  const float*     W2   = (const float*)d_in[6];
  const float*     aS2  = (const float*)d_in[7];
  const float*     aD2  = (const float*)d_in[8];
  const float*     b2   = (const float*)d_in[9];

  const int N  = in_sizes[0] / 256;   // 50000
  const int E  = in_sizes[1] / 2;     // 800000
  const int EE = E + N;

  // workspace carve-up (256B aligned)
  char* w = (char*)d_ws;
  auto take = [&](size_t bytes) { char* p = w; w += (bytes + 255) & ~(size_t)255; return p; };
  __bf16* xb   = (__bf16*)take((size_t)N * 256 * 2);
  __bf16* W1p  = (__bf16*)take((size_t)256 * 512 * 2);   // pre-packed fragments
  __bf16* W2p  = (__bf16*)take((size_t)512 * 64 * 2);    // pre-packed fragments
  float*  h1   = (float*)take((size_t)N * 512 * 4);
  float*  as1  = (float*)take((size_t)N * 8 * 4);
  float*  ad1  = (float*)take((size_t)N * 8 * 4);
  float*  m1   = (float*)take((size_t)N * 8 * 4);
  float*  s1   = (float*)take((size_t)N * 8 * 4);
  float*  alE  = (float*)take((size_t)EE * 8 * 4);       // reused with H=1 in layer 2
  float*  out1 = (float*)take((size_t)N * 512 * 4);
  __bf16* h2b  = (__bf16*)take((size_t)N * 512 * 2);
  float*  g2   = (float*)take((size_t)N * 64 * 4);
  float*  as2  = (float*)take((size_t)N * 4);
  float*  ad2  = (float*)take((size_t)N * 4);
  float*  m2   = (float*)take((size_t)N * 4);
  float*  s2   = (float*)take((size_t)N * 4);

  const float NEG_INF = -__builtin_inff();
  auto blk = [](int n) { return (n + 255) / 256; };

  // ---- precision staging + weight pre-pack ----
  k_convert_bf16<<<blk(N * 256), 256, 0, stream>>>(x, xb, N * 256);
  k_prepack_b<<<blk((256 / 32) * (512 / 16) * 32), 256, 0, stream>>>(W1, W1p, 256, 512);
  k_prepack_b<<<blk((512 / 32) * (64 / 16) * 32), 256, 0, stream>>>(W2, W2p, 512, 64);

  // ---- layer 1 ----
  {
    dim3 g(512 / 128, (N + 127) / 128);
    k_wmma_gemm_direct<128><<<g, 256, 0, stream>>>(xb, W1p, h1, N, 256, 512);
  }
  k_attn<<<(N * 8 + 7) / 8, 256, 0, stream>>>(h1, aS1, aD1, as1, ad1, N, 8);
  k_fill_f32<<<blk(N * 8), 256, 0, stream>>>(m1, NEG_INF, N * 8);
  k_fill_f32<<<blk(N * 8), 256, 0, stream>>>(s1, 0.f, N * 8);
  k_fill_f32<<<blk(N * 512), 256, 0, stream>>>(out1, 0.f, N * 512);
  k_edge_max<<<blk(EE * 8), 256, 0, stream>>>(ei, E, N, 8, as1, ad1, m1);
  k_edge_expsum<<<blk(EE * 8), 256, 0, stream>>>(ei, E, N, 8, as1, ad1, m1, alE, s1);
  k_edge_agg<<<blk(EE * 64), 256, 0, stream>>>(ei, E, N, 8, alE, s1, h1, out1);
  k_relu_bias_bf16<<<blk(N * 512), 256, 0, stream>>>(out1, b1, h2b, N * 512, 512);

  // ---- layer 2 ----
  {
    dim3 g(1, (N + 127) / 128);
    k_wmma_gemm_direct<64><<<g, 256, 0, stream>>>(h2b, W2p, g2, N, 512, 64);
  }
  k_attn<<<(N + 7) / 8, 256, 0, stream>>>(g2, aS2, aD2, as2, ad2, N, 1);
  k_fill_f32<<<blk(N), 256, 0, stream>>>(m2, NEG_INF, N);
  k_fill_f32<<<blk(N), 256, 0, stream>>>(s2, 0.f, N);
  k_init_bias<<<blk(N * 64), 256, 0, stream>>>((float*)d_out, b2, N * 64, 64);
  k_edge_max<<<blk(EE), 256, 0, stream>>>(ei, E, N, 1, as2, ad2, m2);
  k_edge_expsum<<<blk(EE), 256, 0, stream>>>(ei, E, N, 1, as2, ad2, m2, alE, s2);
  k_edge_agg<<<blk(EE * 64), 256, 0, stream>>>(ei, E, N, 1, alE, s2, g2, (float*)d_out);
}
